// mLSTM_79723182948407
// MI455X (gfx1250) — matrix operand order repaired and checked by reading, not compiled, not run on hardware
//
#include <hip/hip_runtime.h>
#include <hip/hip_bf16.h>
#include <stdint.h>

// ---------------------------------------------------------------------------
// Types / CDNA5 helpers
// ---------------------------------------------------------------------------
typedef __bf16 bf16;
typedef __bf16 bf16x16 __attribute__((ext_vector_type(16)));
typedef float  f32x8   __attribute__((ext_vector_type(8)));
typedef int    v4i     __attribute__((vector_size(16))); // 128-bit payload type

#define AS1 __attribute__((address_space(1)))
#define AS3 __attribute__((address_space(3)))

#ifndef __has_builtin
#define __has_builtin(x) 0
#endif
#if __has_builtin(__builtin_amdgcn_global_load_async_to_lds_b128)
#define HAVE_ASYNC_LDS 1
#else
#define HAVE_ASYNC_LDS 0
#endif

__device__ __forceinline__ f32x8 wmma_bf16(bf16x16 a, bf16x16 b, f32x8 c) {
  // D = A(16x32 bf16) x B(32x16 bf16) + C(16x16 f32)   -> v_wmma_f32_16x16x32_bf16
  return __builtin_amdgcn_wmma_f32_16x16x32_bf16(false, a, false, b, (short)0, c,
                                                 false, false);
}

// A-fragment (16x32, bf16): lane<16 -> M=lane, K = {k0..k0+7} U {k0+16..k0+23};
// lanes>=16 shift K by 8 (per ISA 7.12.2 16-bit A layout).
__device__ __forceinline__ bf16x16 frag_a(const bf16* base, int pitch, int row0, int k0) {
  const int lane = threadIdx.x & 31;
  const bf16* p = base + (row0 + (lane & 15)) * pitch + k0 + ((lane >> 4) << 3);
  union { bf16x16 v; uint4 u[2]; } r;
  r.u[0] = *(const uint4*)(p);
  r.u[1] = *(const uint4*)(p + 16);
  return r.v;
}

// B-fragment (32x16, bf16): lane<16 -> N=lane, K = {k0..k0+7} U {k0+16..k0+23};
// lanes>=16 shift K by 8 (same K order as A).
__device__ __forceinline__ bf16x16 frag_b(const bf16* base, int pitch, int col0, int k0) {
  const int lane = threadIdx.x & 31;
  const bf16* p = base + (col0 + (lane & 15)) * pitch + k0 + ((lane >> 4) << 3);
  union { bf16x16 v; uint4 u[2]; } r;
  r.u[0] = *(const uint4*)(p);
  r.u[1] = *(const uint4*)(p + 16);
  return r.v;
}

// 16B global -> LDS copy, async (tracked by ASYNCcnt) on gfx1250.
// Builtin signature (probe-confirmed by diagnostic): param0 = v4i __device__*
// (AS1 global), param1 = LDS pointer, then imm offset + imm cpol.
__device__ __forceinline__ void cp16_g2s(bf16* dst_lds, const bf16* src_g) {
#if HAVE_ASYNC_LDS
  __builtin_amdgcn_global_load_async_to_lds_b128(
      (AS1 v4i*)(uintptr_t)src_g, (AS3 v4i*)(uintptr_t)dst_lds, 0, 0);
#else
  *(uint4*)dst_lds = *(const uint4*)src_g;
#endif
}

__device__ __forceinline__ void cp_wait_barrier() {
#if HAVE_ASYNC_LDS
#if __has_builtin(__builtin_amdgcn_s_wait_asynccnt)
  __builtin_amdgcn_s_wait_asynccnt(0);
#else
  asm volatile("s_wait_asynccnt 0" ::: "memory");
#endif
#endif
  __syncthreads();
}

__device__ __forceinline__ float act_f(float x, int act) {
  switch (act) {
    case 1: return tanhf(x);
    case 2: return 1.f / (1.f + expf(-x));
    case 3: return 0.5f * x * (1.f + erff(x * 0.70710678118654752f)); // exact gelu
  }
  return x;
}

// ---------------------------------------------------------------------------
// Generic bf16 WMMA GEMM:  D[M,N] = epi( A[M,K] @ W[N,K]^T )
// epi: +bias[n], +=Dprev (accum), +resid[m,n], act(), *gate[m,n]; writes f32 and/or bf16
// M % 128 == 0, N % 128 == 0, K % 32 == 0 (all shapes in this model satisfy this)
// ---------------------------------------------------------------------------
__global__ __launch_bounds__(256) void gemm_bf16_wmma_k(
    const bf16* __restrict__ A, const bf16* __restrict__ W,
    const float* __restrict__ bias, const float* __restrict__ resid,
    const float* __restrict__ gate, float* Df, bf16* Dbf,
    int M, int N, int K, int act, int accum)
{
  __shared__ __align__(16) bf16 sA[128 * 40]; // 128 rows x 32 + 8 pad
  __shared__ __align__(16) bf16 sB[128 * 40];
  const int t  = threadIdx.x;
  const int m0 = blockIdx.y << 7;
  const int n0 = blockIdx.x << 7;
  const int wid = t >> 5;
  const int wm = wid >> 2, wn = wid & 3; // 2 x 4 waves -> 64 x 32 per wave

  f32x8 acc[4][2];
  const f32x8 z8 = {0.f,0.f,0.f,0.f,0.f,0.f,0.f,0.f};
#pragma unroll
  for (int i = 0; i < 4; i++)
#pragma unroll
    for (int j = 0; j < 2; j++) acc[i][j] = z8;

  for (int k0 = 0; k0 < K; k0 += 32) {
#pragma unroll
    for (int j = 0; j < 2; j++) {          // 512 16B-vectors per tile, 2 per thread
      int vid = t + (j << 8);
      int row = vid >> 2;
      int kp  = (vid & 3) << 3;
      cp16_g2s(sA + row * 40 + kp, A + (size_t)(m0 + row) * K + k0 + kp);
      cp16_g2s(sB + row * 40 + kp, W + (size_t)(n0 + row) * K + k0 + kp);
    }
    if (k0 + 32 < K) {                     // hint next chunk into cache
      __builtin_prefetch(A + (size_t)(m0 + (t >> 1)) * K + k0 + 32, 0, 1);
      __builtin_prefetch(W + (size_t)(n0 + (t >> 1)) * K + k0 + 32, 0, 1);
    }
    cp_wait_barrier();

    bf16x16 af[4], bfr[2];
#pragma unroll
    for (int mt = 0; mt < 4; mt++) af[mt] = frag_a(sA, 40, wm * 64 + mt * 16, 0);
#pragma unroll
    for (int nt = 0; nt < 2; nt++) bfr[nt] = frag_b(sB, 40, wn * 32 + nt * 16, 0);
#pragma unroll
    for (int mt = 0; mt < 4; mt++)
#pragma unroll
      for (int nt = 0; nt < 2; nt++)
        acc[mt][nt] = wmma_bf16(af[mt], bfr[nt], acc[mt][nt]);
    __syncthreads();
  }

  const int lane = t & 31;
  const int colOff = lane & 15;
  const int rowHalf = (lane >> 4) ? 8 : 0; // D-layout: vgpr g -> row g (+8 for hi half)
#pragma unroll
  for (int mt = 0; mt < 4; mt++)
#pragma unroll
    for (int nt = 0; nt < 2; nt++) {
      const int col = n0 + wn * 32 + nt * 16 + colOff;
      const float bv = bias ? bias[col] : 0.f;
#pragma unroll
      for (int g = 0; g < 8; g++) {
        const int row = m0 + wm * 64 + mt * 16 + g + rowHalf;
        const size_t idx = (size_t)row * N + col;
        float v = acc[mt][nt][g] + bv;
        if (accum) v += Df[idx];
        if (resid) v += resid[idx];
        v = act_f(v, act);
        if (gate) v *= gate[idx];
        if (Df)  Df[idx] = v;
        if (Dbf) Dbf[idx] = (bf16)v;
      }
    }
}

// ---------------------------------------------------------------------------
// Fused per-(batch, head) matrix-memory cell:
//   attn = (q @ k^T) * SCALE ; C = C*f + attn*i ; out = C @ v
// q:[B*D,512] k,v:[B*N,512] bf16 head-chunked; i,f:[B*D, 8*N] f32; C:[B,8,D,N] f32
// ---------------------------------------------------------------------------
__global__ __launch_bounds__(256) void mcell_attn_k(
    const bf16* __restrict__ qb, const bf16* __restrict__ kb, const bf16* __restrict__ vb,
    const float* __restrict__ ig, const float* __restrict__ fg,
    float* C, bf16* outb, int D, int N)
{
  __shared__ __align__(16) bf16 sQ[128 * 72];   // q [D][64] pad8; reused as Cnew [D][N+8]
  __shared__ __align__(16) bf16 sK[128 * 72];   // k [N][64] pad8
  __shared__ __align__(16) bf16 sVT[64 * 136];  // v^T [64][N] pad8
  const int b = blockIdx.x >> 3, h = blockIdx.x & 7;
  const int t = threadIdx.x, lane = t & 31, wid = t >> 5;
  const int colOff = lane & 15, rowHalf = (lane >> 4) ? 8 : 0;
  const f32x8 z8 = {0.f,0.f,0.f,0.f,0.f,0.f,0.f,0.f};

  for (int v = t; v < D * 8; v += 256) {        // q tile
    int r = v >> 3, kp = (v & 7) << 3;
    cp16_g2s(sQ + r * 72 + kp, qb + ((size_t)(b * D + r)) * 512 + h * 64 + kp);
  }
  for (int v = t; v < N * 8; v += 256) {        // k tile
    int r = v >> 3, kp = (v & 7) << 3;
    cp16_g2s(sK + r * 72 + kp, kb + ((size_t)(b * N + r)) * 512 + h * 64 + kp);
  }
  for (int e = t; e < N * 64; e += 256) {       // v^T (scalar transpose)
    int n = e >> 6, kd = e & 63;
    sVT[kd * (N + 8) + n] = vb[((size_t)(b * N + n)) * 512 + h * 64 + kd];
  }
  cp_wait_barrier();

  // Phase 1: attn tiles + gated state update; keep Cnew(bf16) in registers.
  const int tilesN = N >> 4;
  const int nTiles = (D >> 4) * tilesN;
  bf16 creg[4][8];
  int tmArr[4], tnArr[4], cnt = 0;
  for (int tid = wid; tid < nTiles; tid += 8, cnt++) {
    const int tm = tid / tilesN, tn = tid % tilesN;
    f32x8 acc = z8;
    acc = wmma_bf16(frag_a(sQ, 72, tm * 16, 0),  frag_b(sK, 72, tn * 16, 0),  acc);
    acc = wmma_bf16(frag_a(sQ, 72, tm * 16, 32), frag_b(sK, 72, tn * 16, 32), acc);
    const int col = tn * 16 + colOff;
#pragma unroll
    for (int g = 0; g < 8; g++) {
      const int row = tm * 16 + g + rowHalf;
      const size_t ifi = (((size_t)(b * D + row)) * 8 + h) * N + col;
      const size_t ci  = (((size_t)(b * 8 + h)) * D + row) * N + col;
      const float cn = C[ci] * fg[ifi] + acc[g] * 0.125f * ig[ifi]; // SCALE = dk^-0.5
      C[ci] = cn;
      creg[cnt][g] = (bf16)cn;
    }
    tmArr[cnt] = tm; tnArr[cnt] = tn;
  }
  __syncthreads();

  // Phase 2: repack Cnew bf16 into sQ with pitch N+8.
  const int cp = N + 8;
  for (int u = 0; u < cnt; u++) {
    const int col = tnArr[u] * 16 + colOff;
#pragma unroll
    for (int g = 0; g < 8; g++)
      sQ[(tmArr[u] * 16 + g + rowHalf) * cp + col] = creg[u][g];
  }
  __syncthreads();

  // Phase 3: out[D,64] = Cnew[D,N] @ v[N,64]
  const int oTiles = (D >> 4) * 4;
  for (int tid = wid; tid < oTiles; tid += 8) {
    const int tm = tid >> 2, tn = tid & 3;
    f32x8 acc = z8;
    for (int k0 = 0; k0 < N; k0 += 32)
      acc = wmma_bf16(frag_a(sQ, cp, tm * 16, k0), frag_b(sVT, cp, tn * 16, k0), acc);
    const int col = tn * 16 + colOff;
#pragma unroll
    for (int g = 0; g < 8; g++) {
      const int row = tm * 16 + g + rowHalf;
      outb[((size_t)(b * D + row)) * 512 + h * 64 + col] = (bf16)acc[g];
    }
  }
}

// ---------------------------------------------------------------------------
// Elementwise / LN / packing kernels
// ---------------------------------------------------------------------------
__global__ void cvt_f32_bf16_k(const float* __restrict__ s, bf16* __restrict__ d, size_t n) {
  size_t i = (size_t)blockIdx.x * blockDim.x + threadIdx.x;
  const size_t st = (size_t)gridDim.x * blockDim.x;
  for (; i < n; i += st) d[i] = (bf16)s[i];
}

// dst[b*chunk + r] = src[(b*T + t)*chunk + r]  (time-slice + f32->bf16)
__global__ void slice_t_k(const float* __restrict__ s, bf16* __restrict__ d,
                          int t, int T, int chunk, int Btot) {
  size_t i = (size_t)blockIdx.x * blockDim.x + threadIdx.x;
  const size_t n = (size_t)Btot * chunk, st = (size_t)gridDim.x * blockDim.x;
  for (; i < n; i += st) {
    const size_t b = i / chunk, r = i % chunk;
    d[i] = (bf16)s[((size_t)b * T + t) * chunk + r];
  }
}

__global__ void cvt_slice_k(const float* __restrict__ s, bf16* __restrict__ d,
                            int rows, int ld, int off, int cols) {
  size_t i = (size_t)blockIdx.x * blockDim.x + threadIdx.x;
  const size_t n = (size_t)rows * cols, st = (size_t)gridDim.x * blockDim.x;
  for (; i < n; i += st) {
    const size_t r = i / cols, c = i % cols;
    d[i] = (bf16)s[r * ld + off + c];
  }
}

__global__ void sigmoid_slice_k(const float* __restrict__ s, float* __restrict__ d,
                                int rows, int ld, int off, int cols) {
  size_t i = (size_t)blockIdx.x * blockDim.x + threadIdx.x;
  const size_t n = (size_t)rows * cols, st = (size_t)gridDim.x * blockDim.x;
  for (; i < n; i += st) {
    const size_t r = i / cols, c = i % cols;
    d[i] = 1.f / (1.f + expf(-s[r * ld + off + c]));
  }
}

// LayerNorm over 512 columns; optional residual add; writes f32 + bf16.
__global__ __launch_bounds__(256) void ln_k(
    const float* __restrict__ x, const float* __restrict__ add,
    const float* __restrict__ g, const float* __restrict__ b,
    float* __restrict__ of, bf16* __restrict__ obf)
{
  __shared__ float red[256];
  __shared__ float s_mean, s_rstd;
  const int r = blockIdx.x, t = threadIdx.x;
  const size_t base = (size_t)r * 512;
  float v0 = x[base + t], v1 = x[base + t + 256];
  if (add) { v0 += add[base + t]; v1 += add[base + t + 256]; }
  red[t] = v0 + v1; __syncthreads();
  for (int s = 128; s > 0; s >>= 1) { if (t < s) red[t] += red[t + s]; __syncthreads(); }
  if (t == 0) s_mean = red[0] * (1.f / 512.f);
  __syncthreads();
  const float m = s_mean;
  const float d0 = v0 - m, d1 = v1 - m;
  red[t] = d0 * d0 + d1 * d1; __syncthreads();
  for (int s = 128; s > 0; s >>= 1) { if (t < s) red[t] += red[t + s]; __syncthreads(); }
  if (t == 0) s_rstd = rsqrtf(red[0] * (1.f / 512.f) + 1e-5f);
  __syncthreads();
  const float rs = s_rstd;
  const float y0 = d0 * rs * g[t] + b[t];
  const float y1 = d1 * rs * g[t + 256] + b[t + 256];
  of[base + t] = y0; of[base + t + 256] = y1;
  if (obf) { obf[base + t] = (bf16)y0; obf[base + t + 256] = (bf16)y1; }
}

// out[b, 0:128, :] = x1[b] ; out[b, 128:192, :] = x2[b]
__global__ void concat_k(const float* __restrict__ x1, const float* __restrict__ x2,
                         float* __restrict__ out) {
  const size_t n = (size_t)32 * 192 * 512;
  size_t i = (size_t)blockIdx.x * blockDim.x + threadIdx.x;
  const size_t st = (size_t)gridDim.x * blockDim.x;
  for (; i < n; i += st) {
    const int c = (int)(i % 512);
    const size_t bj = i / 512;
    const int j = (int)(bj % 192), b = (int)(bj / 192);
    out[i] = (j < 128) ? x1[((size_t)b * 128 + j) * 512 + c]
                       : x2[((size_t)b * 64 + (j - 128)) * 512 + c];
  }
}

// ---------------------------------------------------------------------------
// Host-side orchestration
// ---------------------------------------------------------------------------
namespace {
struct CellW {  // f32 params (device, from d_in), sorted-key order
  const float *bg, *bfg, *bi, *bk, *bo, *bq, *bv;
  const float *whh, *wih, *wf, *wi, *wk, *wo, *wq, *wv;
};
struct CellB {  // bf16 copies (workspace)
  bf16 *whh, *wih, *wf, *wi, *wk, *wo, *wq, *wv;
};
} // namespace

extern "C" void kernel_launch(void* const* d_in, const int* in_sizes, int n_in,
                              void* d_out, int out_size, void* d_ws, size_t ws_size,
                              hipStream_t stream) {
  (void)in_sizes; (void)n_in; (void)out_size; (void)ws_size;
  constexpr int Bc = 32, Tt = 8, Fc = 128, BSc = 64, Pc = 128, HIDc = 512;
  const int rows1 = Bc * Fc;   // 4096
  const int rows2 = Bc * BSc;  // 2048

  auto in = [&](int i) { return (const float*)d_in[i]; };

  // -------- input mapping (JAX pytree: dict keys sorted) --------
  int pi = 2; // 0:feature 1:base
  const float* feature = in(0);
  const float* basein  = in(1);
  const float *b1b = in(pi++), *b1t = in(pi++), *b2b = in(pi++), *b2t = in(pi++);
  CellW cw[2][2];
  for (int li = 0; li < 2; li++)
    for (int c = 0; c < 2; c++) {
      CellW& w = cw[li][c];
      w.bg = in(pi++); w.bfg = in(pi++); w.bi = in(pi++); w.bk = in(pi++);
      w.bo = in(pi++); w.bq = in(pi++); w.bv = in(pi++);
      w.whh = in(pi++); w.wih = in(pi++); w.wf = in(pi++); w.wi = in(pi++);
      w.wk = in(pi++); w.wo = in(pi++); w.wq = in(pi++); w.wv = in(pi++);
    }
  const float *ln11b = in(pi++), *ln11g = in(pi++), *ln12b = in(pi++), *ln12g = in(pi++);
  const float *ln21b = in(pi++), *ln21g = in(pi++), *ln22b = in(pi++), *ln22g = in(pi++);
  const float *w1b = in(pi++), *w1t = in(pi++), *w2b = in(pi++), *w2t = in(pi++);

  // -------- workspace bump allocator --------
  char* wsbase = (char*)d_ws; size_t off = 0;
  auto alloc = [&](size_t bytes) -> void* {
    void* p = wsbase + off; off = (off + bytes + 255) & ~(size_t)255; return p;
  };
  auto allocF = [&](size_t n) { return (float*)alloc(n * 4); };
  auto allocB = [&](size_t n) { return (bf16*)alloc(n * 2); };

  auto cvt = [&](const float* s, bf16* d, size_t n) {
    int grid = (int)((n + 1023) / 1024); if (grid > 4096) grid = 4096;
    cvt_f32_bf16_k<<<grid, 256, 0, stream>>>(s, d, n);
  };

  // bf16 weight copies
  CellB cb[2][2];
  for (int li = 0; li < 2; li++) {
    const int S = li ? HIDc : Pc;
    for (int c = 0; c < 2; c++) {
      const int NH = (c ? Fc : BSc) * 8;
      CellB& wb = cb[li][c];
      wb.whh = allocB(1536 * 512); wb.wih = allocB((size_t)1536 * S);
      wb.wf = allocB((size_t)NH * 512); wb.wi = allocB((size_t)NH * 512);
      wb.wk = allocB((size_t)512 * S); wb.wo = allocB(512 * 512);
      wb.wq = allocB((size_t)512 * S); wb.wv = allocB((size_t)512 * S);
      const CellW& w = cw[li][c];
      cvt(w.whh, wb.whh, 1536 * 512);       cvt(w.wih, wb.wih, (size_t)1536 * S);
      cvt(w.wf, wb.wf, (size_t)NH * 512);   cvt(w.wi, wb.wi, (size_t)NH * 512);
      cvt(w.wk, wb.wk, (size_t)512 * S);    cvt(w.wo, wb.wo, 512 * 512);
      cvt(w.wq, wb.wq, (size_t)512 * S);    cvt(w.wv, wb.wv, (size_t)512 * S);
    }
  }
  bf16 *w1b_b = allocB(2048 * 512), *w1t_b = allocB(2048 * 512);
  bf16 *w2b_b = allocB(2048 * 512), *w2t_b = allocB(2048 * 512);
  cvt(w1b, w1b_b, 2048 * 512); cvt(w1t, w1t_b, 2048 * 512);
  cvt(w2b, w2b_b, 2048 * 512); cvt(w2t, w2t_b, 2048 * 512);

  // states + activations
  float *h1f[2], *h2f[2], *C1[2], *C2[2]; bf16 *h1b[2], *h2b[2];
  for (int li = 0; li < 2; li++) {
    h1f[li] = allocF((size_t)rows1 * 512); h1b[li] = allocB((size_t)rows1 * 512);
    h2f[li] = allocF((size_t)rows2 * 512); h2b[li] = allocB((size_t)rows2 * 512);
    C1[li] = allocF((size_t)Bc * 8 * Fc * BSc);
    C2[li] = allocF((size_t)Bc * 8 * BSc * Fc);
  }
  float* x1f = allocF((size_t)rows1 * 512); bf16* x1b = allocB((size_t)rows1 * 512);
  float* x2f = allocF((size_t)rows2 * 512); bf16* x2b = allocB((size_t)rows2 * 512);
  bf16* fx1 = allocB((size_t)rows1 * Pc);
  bf16* fx2 = allocB((size_t)rows2 * Pc);
  float* gates = allocF((size_t)rows1 * 1536);
  bf16 *igb = allocB((size_t)rows1 * 512), *fgb = allocB((size_t)rows1 * 512);
  float* ofg = allocF((size_t)rows1 * 512);
  float *i_f = allocF((size_t)rows1 * 1024), *f_f = allocF((size_t)rows1 * 1024);
  bf16 *qb = allocB((size_t)rows1 * 512), *kb = allocB((size_t)rows1 * 512);
  bf16 *vb = allocB((size_t)rows1 * 512), *aob = allocB((size_t)rows1 * 512);
  bf16* midb = allocB((size_t)rows1 * 2048);
  float* ftmp = allocF((size_t)rows1 * 512);

  // zero recurrent state every call (deterministic)
  for (int li = 0; li < 2; li++) {
    (void)hipMemsetAsync(h1f[li], 0, (size_t)rows1 * 512 * 4, stream);
    (void)hipMemsetAsync(h1b[li], 0, (size_t)rows1 * 512 * 2, stream);
    (void)hipMemsetAsync(h2f[li], 0, (size_t)rows2 * 512 * 4, stream);
    (void)hipMemsetAsync(h2b[li], 0, (size_t)rows2 * 512 * 2, stream);
    (void)hipMemsetAsync(C1[li], 0, (size_t)Bc * 8 * Fc * BSc * 4, stream);
    (void)hipMemsetAsync(C2[li], 0, (size_t)Bc * 8 * BSc * Fc * 4, stream);
  }

  auto gemm = [&](const bf16* A, const bf16* W, const float* bias, const float* resid,
                  const float* gate, float* Df, bf16* Dbf, int M, int N, int K,
                  int act, int accum) {
    gemm_bf16_wmma_k<<<dim3(N / 128, M / 128), 256, 0, stream>>>(
        A, W, bias, resid, gate, Df, Dbf, M, N, K, act, accum);
  };
  auto ln = [&](const float* x, const float* add, const float* g, const float* b,
                float* of2, bf16* ob2, int rows) {
    ln_k<<<rows, 256, 0, stream>>>(x, add, g, b, of2, ob2);
  };

  auto run_cell = [&](const CellW& w, const CellB& wb, int S, int rowsA, int rowsB,
                      int Nn, const bf16* xinA, const bf16* xinB,
                      float* hf, bf16* hb, float* Cst) {
    const int D = rowsA / Bc, NH = Nn * 8;
    // gates = xA @ w_ih^T + b_g + h @ w_hh^T
    gemm(xinA, wb.wih, w.bg, nullptr, nullptr, gates, nullptr, rowsA, 1536, S, 0, 0);
    gemm(hb,   wb.whh, nullptr, nullptr, nullptr, gates, nullptr, rowsA, 1536, 512, 0, 1);
    {
      const size_t n = (size_t)rowsA * 512; const int grid = (int)((n + 1023) / 1024);
      cvt_slice_k<<<grid, 256, 0, stream>>>(gates, igb, rowsA, 1536, 0, 512);
      cvt_slice_k<<<grid, 256, 0, stream>>>(gates, fgb, rowsA, 1536, 512, 512);
      sigmoid_slice_k<<<grid, 256, 0, stream>>>(gates, ofg, rowsA, 1536, 1024, 512);
    }
    gemm(igb, wb.wi, w.bi, nullptr, nullptr, i_f, nullptr, rowsA, NH, 512, 1, 0); // tanh
    gemm(fgb, wb.wf, w.bfg, nullptr, nullptr, f_f, nullptr, rowsA, NH, 512, 2, 0); // sigm
    gemm(xinA, wb.wq, w.bq, nullptr, nullptr, nullptr, qb, rowsA, 512, S, 0, 0);
    gemm(xinB, wb.wk, w.bk, nullptr, nullptr, nullptr, kb, rowsB, 512, S, 0, 0);
    gemm(xinB, wb.wv, w.bv, nullptr, nullptr, nullptr, vb, rowsB, 512, S, 0, 0);
    mcell_attn_k<<<Bc * 8, 256, 0, stream>>>(qb, kb, vb, i_f, f_f, Cst, aob, D, Nn);
    // h = sigmoid(og) * (out @ wo^T + bo)
    gemm(aob, wb.wo, w.bo, nullptr, ofg, hf, hb, rowsA, 512, 512, 0, 0);
  };

  // -------- time loop --------
  for (int t = 0; t < Tt; t++) {
    slice_t_k<<<2048, 256, 0, stream>>>(feature, fx1, t, Tt, Fc * Pc, Bc);
    slice_t_k<<<1024, 256, 0, stream>>>(basein, fx2, t, Tt, BSc * Pc, Bc);
    const bf16* xin1 = fx1; const bf16* xin2 = fx2; int S = Pc;
    for (int li = 0; li < 2; li++) {
      run_cell(cw[li][0], cb[li][0], S, rows1, rows2, BSc, xin1, xin2,
               h1f[li], h1b[li], C1[li]);
      run_cell(cw[li][1], cb[li][1], S, rows2, rows1, Fc, xin2, xin1,
               h2f[li], h2b[li], C2[li]);
      // stream 1: (x1 [+]= h1) -> LN11 -> +FFN_b -> LN12
      ln(li ? x1f : h1f[li], li ? h1f[li] : nullptr, ln11g, ln11b, x1f, x1b, rows1);
      gemm(x1b, w1b_b, b1b, nullptr, nullptr, nullptr, midb, rows1, 2048, 512, 3, 0);
      gemm(midb, w2b_b, b2b, x1f, nullptr, ftmp, nullptr, rows1, 512, 2048, 0, 0);
      ln(ftmp, nullptr, ln12g, ln12b, x1f, x1b, rows1);
      // stream 2: (x2 [+]= h2) -> LN21 -> +FFN_t -> LN22
      ln(li ? x2f : h2f[li], li ? h2f[li] : nullptr, ln21g, ln21b, x2f, x2b, rows2);
      gemm(x2b, w1t_b, b1t, nullptr, nullptr, nullptr, midb, rows2, 2048, 512, 3, 0);
      gemm(midb, w2t_b, b2t, x2f, nullptr, ftmp, nullptr, rows2, 512, 2048, 0, 0);
      ln(ftmp, nullptr, ln22g, ln22b, x2f, x2b, rows2);
      xin1 = x1b; xin2 = x2b; S = HIDc;
    }
  }
  concat_k<<<4096, 256, 0, stream>>>(x1f, x2f, (float*)d_out);
}
